// MLPAttention_15788299780764
// MI455X (gfx1250) — matrix operand order repaired and checked by reading, not compiled, not run on hardware
//
#include <hip/hip_runtime.h>
#include <hip/hip_bf16.h>
#include <math.h>

typedef __attribute__((ext_vector_type(16))) _Float16 v16h;
typedef __attribute__((ext_vector_type(8)))  float    v8f;

#define BB 4
#define NQ 512
#define NK 512
#define DD 64
#define TAUF 4.0f

// ---------------------------------------------------------------------------
// Kernel 1: hq = Q @ W1[0::2],  hkb = K @ W1[1::2] + b1  (b1 folded into hk
// so the hot scoring loop does one fewer add per element).
// One block (64 threads) per output row; row staged in LDS, thread e owns col e.
// ---------------------------------------------------------------------------
__global__ void __launch_bounds__(64)
proj_kernel(const float* __restrict__ Q, const float* __restrict__ Kin,
            const float* __restrict__ W1, const float* __restrict__ b1,
            float* __restrict__ hq, float* __restrict__ hkb) {
  __shared__ float s_row[DD];
  int r = blockIdx.x;
  int e = threadIdx.x;                 // 0..63
  const float* src; float* dst; int sel; float bias;
  if (r < BB * NQ) { src = Q + (size_t)r * DD;    dst = hq  + (size_t)r * DD; sel = 0; bias = 0.f; }
  else             { int r2 = r - BB * NQ;
                     src = Kin + (size_t)r2 * DD; dst = hkb + (size_t)r2 * DD; sel = 1; bias = b1[e]; }
  s_row[e] = src[e];
  __syncthreads();
  float acc = bias;
  #pragma unroll 8
  for (int d = 0; d < DD; ++d)
    acc = fmaf(s_row[d], W1[(2 * d + sel) * DD + e], acc);
  dst[e] = acc;
}

// ---------------------------------------------------------------------------
// Kernel 2: Vt[b][d][k] = (f16) V[b][k][d]  -- so WMMA B-operand columns are
// contiguous 16-half segments in memory.
// ---------------------------------------------------------------------------
__global__ void __launch_bounds__(256)
vt_kernel(const float* __restrict__ V, _Float16* __restrict__ Vt) {
  int i = blockIdx.x * 256 + threadIdx.x;      // < B*D*K
  int b   = i / (DD * NK);
  int rem = i % (DD * NK);
  int d   = rem / NK;
  int k   = rem % NK;
  Vt[i] = (_Float16)V[((size_t)(b * NK + k)) * DD + d];
}

// ---------------------------------------------------------------------------
// Kernel 3: per (batch, 16-query tile), 256 threads = 8 waves:
//   VALU relu-MLP scores (dominant cost) -> parallel softmax -> att out (f32)
//   -> PV via v_wmma_f32_16x16x32_f16 on waves 0..3.
// ---------------------------------------------------------------------------
__global__ void __launch_bounds__(256)
attn_kernel(const float* __restrict__ hq, const float* __restrict__ hkb,
            const _Float16* __restrict__ Vt,
            const float* __restrict__ W2,
            float* __restrict__ out_g, float* __restrict__ att_g) {
  __shared__ float    s_hq[16 * DD];        // 4 KB
  __shared__ float    s_hk[32 * DD];        // 8 KB
  __shared__ float    s_res[16 * NK];       // 32 KB
  __shared__ _Float16 s_att[16 * NK];       // 16 KB
  __shared__ float    s_w2[DD];
  __shared__ float    s_red[256];           // softmax tree scratch
  __shared__ float    s_inv[16];

  const int tid   = threadIdx.x;
  const int b     = blockIdx.x >> 5;        // / (NQ/16)
  const int ntile = blockIdx.x & 31;
  const int n0    = ntile * 16;

  // W2 pre-scaled by 1/TAU (softmax scale folded in); b2 dropped (shift-inv.)
  if (tid < DD) s_w2[tid] = W2[tid] * (1.0f / TAUF);
  const float* hq_base = hq + ((size_t)(b * NQ + n0)) * DD;
  for (int i = tid; i < 16 * DD; i += 256) s_hq[i] = hq_base[i];
  __syncthreads();

  // ---- scores: res[m][k] = (1/TAU) * sum_d relu(hq + hkb) * W2 -------------
  // Thread -> (row m, columns kk and kk+16): hq read once feeds 2 FMAs.
  const int m   = tid >> 4;                 // 0..15
  const int kk0 = tid & 15;                 // 0..15
  for (int kt = 0; kt < NK / 32; ++kt) {
    const float* hk_base = hkb + ((size_t)(b * NK + kt * 32)) * DD;
    for (int i = tid; i < 32 * DD; i += 256) s_hk[i] = hk_base[i];
    if (kt + 1 < NK / 32)                    // global_prefetch_b8 of next tile
      __builtin_prefetch(hkb + ((size_t)(b * NK + (kt + 1) * 32)) * DD + tid * 8, 0, 0);
    __syncthreads();
    {
      const float* hqr = &s_hq[m * DD];
      const float* hk0 = &s_hk[kk0 * DD];
      const float* hk1 = &s_hk[(kk0 + 16) * DD];
      float acc0 = 0.f, acc1 = 0.f;
      #pragma unroll 8
      for (int d = 0; d < DD; ++d) {
        float q = hqr[d];
        float w = s_w2[d];
        acc0 = fmaf(fmaxf(q + hk0[d], 0.f), w, acc0);
        acc1 = fmaf(fmaxf(q + hk1[d], 0.f), w, acc1);
      }
      s_res[m * NK + kt * 32 + kk0]      = acc0;
      s_res[m * NK + kt * 32 + kk0 + 16] = acc1;
    }
    __syncthreads();
  }

  // ---- softmax over k: 16 threads per row, LDS tree reduction -------------
  {
    const int row = tid >> 4, seg = tid & 15;
    const float* r = &s_res[row * NK + seg * 32];
    float mx = -INFINITY;
    #pragma unroll 8
    for (int k = 0; k < 32; ++k) mx = fmaxf(mx, r[k]);
    s_red[tid] = mx;
  }
  __syncthreads();
  float rowmax;
  {
    const int row = tid >> 4;
    float mx = -INFINITY;
    #pragma unroll
    for (int j = 0; j < 16; ++j) mx = fmaxf(mx, s_red[row * 16 + j]);
    rowmax = mx;
  }
  __syncthreads();
  {
    const int row = tid >> 4, seg = tid & 15;
    float* r = &s_res[row * NK + seg * 32];
    float sum = 0.f;
    #pragma unroll 8
    for (int k = 0; k < 32; ++k) { float e = __expf(r[k] - rowmax); r[k] = e; sum += e; }
    s_red[tid] = sum;
  }
  __syncthreads();
  if ((tid & 15) == 0) {
    const int row = tid >> 4;
    float s = 0.f;
    #pragma unroll
    for (int j = 0; j < 16; ++j) s += s_red[row * 16 + j];
    s_inv[row] = 1.0f / s;
  }
  __syncthreads();

  // ---- normalize, emit att (f32 output #2), stash f16 copy for WMMA ------
  float* att_base = att_g + ((size_t)(b * NQ + n0)) * NK;
  for (int i = tid; i < 16 * NK; i += 256) {
    int row = i >> 9;                     // / NK
    float v = s_res[i] * s_inv[row];
    att_base[i] = v;
    s_att[i]    = (_Float16)v;
  }
  __syncthreads();

  // ---- out = att @ V via v_wmma_f32_16x16x32_f16 (waves 0..3) ------------
  const int wid  = tid >> 5;              // wave id (uniform per wave)
  const int lane = tid & 31;
  if (wid < 4) {
    const int mm = lane & 15;             // A row / B column / D column
    const int h  = lane >> 4;             // half select per ISA layouts
    v8f c = {0.f, 0.f, 0.f, 0.f, 0.f, 0.f, 0.f, 0.f};
    const _Float16* vt_base = Vt + ((size_t)(b * DD + wid * 16 + mm)) * NK;
    for (int kt = 0; kt < NK / 32; ++kt) {
      v16h a, bm;
      // A: 16x32 f16 layout: lane mm, halves j -> K = 16*h + j
      const _Float16* ar = &s_att[mm * NK + kt * 32];
      const int kb = h * 16;
      #pragma unroll
      for (int j = 0; j < 16; ++j) { a[j] = ar[kb + j]; }
      // B: 32x16 f16 layout: lane holds column mm, K = 16*h + j (contiguous in Vt)
      const _Float16* br = vt_base + kt * 32 + h * 16;
      #pragma unroll
      for (int j = 0; j < 16; ++j) bm[j] = br[j];
      c = __builtin_amdgcn_wmma_f32_16x16x32_f16(
          /*neg_a=*/false, a, /*neg_b=*/false, bm,
          /*c_mod=*/(short)0, c, /*reuse_a=*/false, /*reuse_b=*/false);
    }
    // D: VGPR r -> row = r + 8*h, col = mm
    float* out_base = out_g + ((size_t)(b * NQ + n0)) * DD + wid * 16;
    #pragma unroll
    for (int r = 0; r < 8; ++r) {
      int row = r + h * 8;
      out_base[(size_t)row * DD + mm] = c[r];
    }
  }
}

// ---------------------------------------------------------------------------
extern "C" void kernel_launch(void* const* d_in, const int* in_sizes, int n_in,
                              void* d_out, int out_size, void* d_ws, size_t ws_size,
                              hipStream_t stream) {
  const float* Q   = (const float*)d_in[0];
  const float* Kin = (const float*)d_in[1];
  const float* V   = (const float*)d_in[2];
  const float* W1  = (const float*)d_in[3];
  const float* b1  = (const float*)d_in[4];
  const float* W2  = (const float*)d_in[5];

  float* out_g = (float*)d_out;                         // B*N*D
  float* att_g = (float*)d_out + (size_t)BB * NQ * DD;  // B*N*K

  float*    hq  = (float*)d_ws;                           // B*N*D f32
  float*    hkb = hq + (size_t)BB * NQ * DD;              // B*K*D f32 (+b1)
  _Float16* Vt  = (_Float16*)(hkb + (size_t)BB * NK * DD);// B*D*K f16

  proj_kernel<<<BB * (NQ + NK), 64, 0, stream>>>(Q, Kin, W1, b1, hq, hkb);
  vt_kernel<<<(BB * DD * NK) / 256, 256, 0, stream>>>(V, Vt);
  attn_kernel<<<BB * (NQ / 16), 256, 0, stream>>>(hq, hkb, Vt, W2,
                                                  out_g, att_g);
}
